// YourModel_37340445672196
// MI455X (gfx1250) — matrix operand order repaired and checked by reading, not compiled, not run on hardware
//
#include <hip/hip_runtime.h>
#include <hip/hip_bf16.h>
#include <cstdint>

// ---------------------------------------------------------------------------
// Types for CDNA5 WMMA (wave32)
// ---------------------------------------------------------------------------
typedef __attribute__((ext_vector_type(16))) __bf16 v16bf;
typedef __attribute__((ext_vector_type(8)))  __bf16 v8bf;
typedef __attribute__((ext_vector_type(8)))  float  v8f;

// ---------------------------------------------------------------------------
// GEMM: Y[M,N] = X[M,K](bf16,row-major) @ W[N,K](bf16,row-major)^T + bias
// Block (256 thr, 8 waves) computes a 128(M) x 64(N) macro-tile; each wave a
// 16x64 tile of it. The shared 64x32 B-panel per k-step is staged into LDS
// with double-buffered global_load_async_to_lds_b128 (ASYNCcnt), overlapped
// with v_wmma_f32_16x16x32_bf16 on the previous buffer.
// Requires M % 128 == 0, K % 32 == 0. N may be ragged (block-uniform edge path).
// ---------------------------------------------------------------------------
#define BROW_STRIDE 80   // bytes per LDS row (64 data + 16 pad, 16B aligned)
#define BUF_BYTES   (64 * BROW_STRIDE)

__global__ __launch_bounds__(256) void gemm_bf16_kernel(
    const __bf16* __restrict__ X, const __bf16* __restrict__ W,
    const float* __restrict__ bias, float* __restrict__ Y,
    int M, int N, int K)
{
    __shared__ __align__(16) char smem[2][BUF_BYTES];
    const int lane = threadIdx.x & 31;
    const int wave = threadIdx.x >> 5;
    const int kh   = lane >> 4;      // half-wave select (0/1)
    const int l16  = lane & 15;
    const int n0 = blockIdx.x << 6;
    const int m0 = (blockIdx.y << 7) + (wave << 4);

    v8f acc[4] = {};
    const __bf16* xrow = X + (size_t)(m0 + l16) * K;

    if (n0 + 64 <= N) {
        // ---------------- fast path: full tile, LDS-staged B ----------------
        const uint32_t sbase = (uint32_t)(uintptr_t)(&smem[0][0]);
        const int r = threadIdx.x >> 2;   // B row 0..63 handled by this thread
        const int c = threadIdx.x & 3;    // 16-byte chunk 0..3 within the row
        const __bf16* wsrc = W + (size_t)(n0 + r) * K + c * 8;
        const uint32_t sdst = sbase + r * BROW_STRIDE + c * 16;

        // prologue: stage k-chunk 0 into buffer 0
        asm volatile("global_load_async_to_lds_b128 %0, %1, off"
                     :: "v"(sdst), "v"(wsrc) : "memory");

        const int nChunks = K >> 5;
        for (int kc = 0; kc < nChunks; ++kc) {
            const int cur = kc & 1;
            asm volatile("s_wait_asynccnt 0x0" ::: "memory");
            __syncthreads();  // current buffer complete across all waves
            if (kc + 1 < nChunks) {
                const __bf16* src = wsrc + (size_t)(kc + 1) * 32;
                const uint32_t dst = sdst + (1 - cur) * BUF_BYTES;
                asm volatile("global_load_async_to_lds_b128 %0, %1, off"
                             :: "v"(dst), "v"(src) : "memory");
            }
            const int k0 = kc << 5;
            // A fragment 16x32: elems 0..15 = A[m][k0+16*kh..k0+16*kh+15]
            union { v16bf v; v8bf h[2]; } A;
            A.h[0] = *(const v8bf*)(xrow + k0 + 16 * kh);
            A.h[1] = *(const v8bf*)(xrow + k0 + 16 * kh + 8);
            const char* bufp = &smem[cur][0];
#pragma unroll
            for (int nj = 0; nj < 4; ++nj) {
                // B fragment 32x16: lane(kh, n=l16) needs W-row (nj*16+l16),
                // k-bytes 32*kh .. 32*kh+31 within the staged row
                const char* bp = bufp + (nj * 16 + l16) * BROW_STRIDE + 32 * kh;
                union { v16bf v; v8bf h[2]; } Bf;
                Bf.h[0] = *(const v8bf*)(bp);
                Bf.h[1] = *(const v8bf*)(bp + 16);
                acc[nj] = __builtin_amdgcn_wmma_f32_16x16x32_bf16(
                    false, A.v, false, Bf.v, (short)0, acc[nj], false, false);
            }
        }
    } else {
        // ---------------- ragged-N edge: direct global, per-lane guard ------
        const v16bf zfrag = {};
        for (int k0 = 0; k0 < K; k0 += 32) {
            union { v16bf v; v8bf h[2]; } A;
            A.h[0] = *(const v8bf*)(xrow + k0 + 16 * kh);
            A.h[1] = *(const v8bf*)(xrow + k0 + 16 * kh + 8);
#pragma unroll
            for (int nj = 0; nj < 4; ++nj) {
                const int n = n0 + nj * 16 + l16;
                union { v16bf v; v8bf h[2]; } Bf;
                if (n < N) {
                    const __bf16* wrow = W + (size_t)n * K + k0 + 16 * kh;
                    Bf.h[0] = *(const v8bf*)(wrow);
                    Bf.h[1] = *(const v8bf*)(wrow + 8);
                } else {
                    Bf.v = zfrag;
                }
                acc[nj] = __builtin_amdgcn_wmma_f32_16x16x32_bf16(
                    false, A.v, false, Bf.v, (short)0, acc[nj], false, false);
            }
        }
    }

    // C/D layout: lane col = l16, VGPR v -> row v + 8*kh
#pragma unroll
    for (int nj = 0; nj < 4; ++nj) {
        const int n = n0 + nj * 16 + l16;
        if (n < N) {
            const float bv = bias[n];
#pragma unroll
            for (int v = 0; v < 8; ++v)
                Y[(size_t)(m0 + v + 8 * kh) * N + n] = acc[nj][v] + bv;
        }
    }
}

// ---------------------------------------------------------------------------
// Elementwise / data-movement kernels
// ---------------------------------------------------------------------------
__global__ void fill_f32_kernel(float* p, float v, size_t n) {
    size_t i = (size_t)blockIdx.x * blockDim.x + threadIdx.x;
    if (i < n) p[i] = v;
}

__global__ void cvt_bf16_kernel(const float* __restrict__ in, __bf16* __restrict__ out, size_t n) {
    size_t i = (size_t)blockIdx.x * blockDim.x + threadIdx.x;
    if (i < n) out[i] = (__bf16)in[i];
}

// out[n*K + k] = in[k*N + n]   (in is K x N row-major)
__global__ void transpose_cvt_kernel(const float* __restrict__ in, __bf16* __restrict__ out,
                                     int K, int N) {
    size_t i = (size_t)blockIdx.x * blockDim.x + threadIdx.x;
    if (i >= (size_t)K * N) return;
    int k = (int)(i / N), n = (int)(i % N);
    out[(size_t)n * K + k] = (__bf16)in[i];
}

// xcB[(t*512+b)*768 + k] = bf16(emb_city[t][x[b,t]][k])
__global__ void gather_xc_kernel(const int* __restrict__ x, const float* __restrict__ emb_city,
                                 __bf16* __restrict__ xcB) {
    size_t i = (size_t)blockIdx.x * blockDim.x + threadIdx.x;
    if (i >= (size_t)5 * 512 * 768) return;
    int k = (int)(i % 768);
    size_t r = i / 768;
    int b = (int)(r % 512);
    int t = (int)(r / 512);
    int city = x[b * 11 + t];
    xcB[i] = (__bf16)emb_city[((size_t)t * 39000 + city) * 768 + k];
}

__device__ __forceinline__ float sigmoidf_(float v) { return 1.f / (1.f + __expf(-v)); }

__global__ void gru_gate_kernel(const float* __restrict__ gi, const float* __restrict__ gh,
                                float* __restrict__ hF, __bf16* __restrict__ hB) {
    size_t i = (size_t)blockIdx.x * blockDim.x + threadIdx.x;
    if (i >= (size_t)512 * 768) return;
    int b = (int)(i / 768), c = (int)(i % 768);
    size_t base = (size_t)b * 2304;
    float ir = gi[base + c], iz = gi[base + 768 + c], in = gi[base + 1536 + c];
    float hr = gh[base + c], hz = gh[base + 768 + c], hn = gh[base + 1536 + c];
    float r = sigmoidf_(ir + hr);
    float z = sigmoidf_(iz + hz);
    float n = tanhf(in + r * hn);
    float h = (1.f - z) * n + z * hF[i];
    hF[i] = h;
    hB[i] = (__bf16)h;
}

__global__ void feat_kernel(const int* __restrict__ x,
                            const float* __restrict__ e0, const float* __restrict__ e1,
                            const float* __restrict__ e2, const float* __restrict__ e3,
                            const float* __restrict__ e4, const float* __restrict__ e5,
                            const float* __restrict__ hF, __bf16* __restrict__ featB) {
    size_t i = (size_t)blockIdx.x * blockDim.x + threadIdx.x;
    if (i >= (size_t)512 * 896) return;
    int b = (int)(i / 896), c = (int)(i % 896);
    float v;
    if (c < 128) {
        const int* xb = x + b * 11 + 5;
        if      (c < 16)  v = e0[xb[0] * 16 + c];
        else if (c < 32)  v = e1[xb[1] * 16 + (c - 16)];
        else if (c < 64)  v = e2[xb[2] * 32 + (c - 32)];
        else if (c < 96)  v = e3[xb[3] * 32 + (c - 64)];
        else if (c < 112) v = e4[xb[4] * 16 + (c - 96)];
        else              v = e5[xb[5] * 16 + (c - 112)];
    } else {
        v = hF[(size_t)b * 768 + (c - 128)];
    }
    featB[i] = (__bf16)v;
}

// BatchNorm over axis 0 (M=512 rows) + ReLU -> bf16. One wave per column (wave32).
__global__ void bn_relu_kernel(const float* __restrict__ Xin, const float* __restrict__ g,
                               const float* __restrict__ b, __bf16* __restrict__ outB,
                               int Ncols) {
    const int lane = threadIdx.x & 31;
    const int col  = blockIdx.x * (blockDim.x >> 5) + (threadIdx.x >> 5);
    if (col >= Ncols) return;
    float vals[16];
    float s = 0.f, s2 = 0.f;
#pragma unroll
    for (int i = 0; i < 16; ++i) {
        float v = Xin[(size_t)(lane + i * 32) * Ncols + col];
        vals[i] = v; s += v; s2 += v * v;
    }
#pragma unroll
    for (int m = 16; m >= 1; m >>= 1) {
        s  += __shfl_xor(s,  m, 32);
        s2 += __shfl_xor(s2, m, 32);
    }
    float mean = s * (1.f / 512.f);
    float var  = s2 * (1.f / 512.f) - mean * mean;
    float inv  = rsqrtf(var + 1e-5f);
    float gg = g[col], bb = b[col];
#pragma unroll
    for (int i = 0; i < 16; ++i) {
        float y = gg * (vals[i] - mean) * inv + bb;
        outB[(size_t)(lane + i * 32) * Ncols + col] = (__bf16)(y > 0.f ? y : 0.f);
    }
}

// Top-50 per row, no mutation: each round selects the best (value desc, idx asc)
// element strictly after the previously selected key.
__global__ __launch_bounds__(256) void topk_kernel(const float* __restrict__ Prob,
                                                   int* __restrict__ topId, int T) {
    const int b = blockIdx.x;
    const int tid = threadIdx.x;
    __shared__ float sv[256];
    __shared__ int   si[256];
    __shared__ float lastV_s;
    __shared__ int   lastI_s;
    if (tid == 0) { lastV_s = 3.4e38f; lastI_s = -1; }
    __syncthreads();
    const float* row = Prob + (size_t)b * T;
    for (int sel = 0; sel < 50; ++sel) {
        float lastV = lastV_s; int lastI = lastI_s;
        float bv = -3.4e38f; int bi = T;
        for (int j = tid; j < T; j += 256) {
            float v = row[j];
            bool after = (v < lastV) || (v == lastV && j > lastI);
            if (after && (v > bv || (v == bv && j < bi))) { bv = v; bi = j; }
        }
        sv[tid] = bv; si[tid] = bi;
        __syncthreads();
        for (int off = 128; off > 0; off >>= 1) {
            if (tid < off) {
                float ov = sv[tid + off]; int oi = si[tid + off];
                if (ov > sv[tid] || (ov == sv[tid] && oi < si[tid])) { sv[tid] = ov; si[tid] = oi; }
            }
            __syncthreads();
        }
        if (tid == 0) { topId[b * 50 + sel] = si[0]; lastV_s = sv[0]; lastI_s = si[0]; }
        __syncthreads();
    }
}

// s[b,n] = e[b,:].dot(emb0[topId[b,n],:]); softmax over n; scatter p+1e-6.
__global__ void emb_dot_kernel(const float* __restrict__ E, const float* __restrict__ emb0,
                               const int* __restrict__ topId, float* __restrict__ Out, int T) {
    const int b = blockIdx.x;
    const int tid = threadIdx.x;  // 64 threads
    __shared__ float s[50];
    __shared__ float red[2];
    if (tid < 50) {
        int id = topId[b * 50 + tid];
        const float* e  = E + (size_t)b * 768;
        const float* te = emb0 + (size_t)id * 768;
        float acc = 0.f;
        for (int k = 0; k < 768; ++k) acc += e[k] * te[k];
        s[tid] = acc;
    }
    __syncthreads();
    if (tid == 0) {
        float m = -3.4e38f;
        for (int i = 0; i < 50; ++i) m = fmaxf(m, s[i]);
        red[0] = m;
    }
    __syncthreads();
    if (tid < 50) s[tid] = __expf(s[tid] - red[0]);
    __syncthreads();
    if (tid == 0) {
        float t = 0.f;
        for (int i = 0; i < 50; ++i) t += s[i];
        red[1] = t;
    }
    __syncthreads();
    if (tid < 50) {
        int id = topId[b * 50 + tid];
        Out[(size_t)b * T + id] = s[tid] / red[1] + 1e-6f;
    }
}

__global__ void combine_kernel(const float* __restrict__ prob, const float* __restrict__ p1,
                               const float* __restrict__ p2, const float* __restrict__ w1p,
                               const float* __restrict__ w2p, float* __restrict__ pw, size_t n) {
    size_t i = (size_t)blockIdx.x * blockDim.x + threadIdx.x;
    if (i >= n) return;
    float w1s = 1.f / (1.f + __expf(-w1p[0]));
    float w2s = 1.f / (1.f + __expf(-w2p[0]));
    pw[i] = (prob[i] + p1[i] * w1s + p2[i] * w2s) / (1.f + w1s + w2s);
}

// ---------------------------------------------------------------------------
// Host-side orchestration
// ---------------------------------------------------------------------------
extern "C" void kernel_launch(void* const* d_in, const int* in_sizes, int n_in,
                              void* d_out, int out_size, void* d_ws, size_t ws_size,
                              hipStream_t stream) {
    (void)in_sizes; (void)n_in; (void)out_size; (void)ws_size;
    const int*   x        = (const int*)  d_in[0];
    const float* emb_city = (const float*)d_in[1];
    const float* e0  = (const float*)d_in[2];
    const float* e1  = (const float*)d_in[3];
    const float* e2  = (const float*)d_in[4];
    const float* e3  = (const float*)d_in[5];
    const float* e4  = (const float*)d_in[6];
    const float* e5  = (const float*)d_in[7];
    const float* Wih = (const float*)d_in[8];
    const float* Whh = (const float*)d_in[9];
    const float* bih = (const float*)d_in[10];
    const float* bhh = (const float*)d_in[11];
    const float* Wc  = (const float*)d_in[12];
    const float* bc  = (const float*)d_in[13];
    const float* W1  = (const float*)d_in[14];
    const float* b1  = (const float*)d_in[15];
    const float* g1  = (const float*)d_in[16];
    const float* be1 = (const float*)d_in[17];
    const float* W2  = (const float*)d_in[18];
    const float* b2  = (const float*)d_in[19];
    const float* g2  = (const float*)d_in[20];
    const float* be2 = (const float*)d_in[21];
    const float* Wp  = (const float*)d_in[22];
    const float* bp  = (const float*)d_in[23];
    const float* Wd1 = (const float*)d_in[24];
    const float* bd1 = (const float*)d_in[25];
    const float* Wd2 = (const float*)d_in[26];
    const float* bd2 = (const float*)d_in[27];
    const float* w1p = (const float*)d_in[28];
    const float* w2p = (const float*)d_in[29];
    float* out = (float*)d_out;

    constexpr int B = 512, T = 39000, EC = 768, H = 768, CIN = 896, G3 = 2304;
    constexpr size_t S = (size_t)B * T;

    // workspace carve-up
    char* wsp = (char*)d_ws;
    auto alloc = [&](size_t bytes) -> void* {
        void* r = (void*)wsp;
        wsp += (bytes + 255) & ~(size_t)255;
        return r;
    };
    __bf16* WihB  = (__bf16*)alloc((size_t)G3 * EC * 2);
    __bf16* WhhB  = (__bf16*)alloc((size_t)G3 * EC * 2);
    __bf16* WcT   = (__bf16*)alloc((size_t)H * CIN * 2);
    __bf16* W1T   = (__bf16*)alloc((size_t)H * H * 2);
    __bf16* W2T   = (__bf16*)alloc((size_t)H * H * 2);
    __bf16* Wd1T  = (__bf16*)alloc((size_t)H * H * 2);
    __bf16* Wd2T  = (__bf16*)alloc((size_t)H * H * 2);
    __bf16* WpT   = (__bf16*)alloc((size_t)T * H * 2);
    __bf16* xcB   = (__bf16*)alloc((size_t)5 * B * EC * 2);
    float*  hF    = (float*) alloc((size_t)B * EC * 4);
    __bf16* hB    = (__bf16*)alloc((size_t)B * EC * 2);
    float*  gi    = (float*) alloc((size_t)B * G3 * 4);
    float*  gh    = (float*) alloc((size_t)B * G3 * 4);
    __bf16* featB = (__bf16*)alloc((size_t)B * CIN * 2);
    float*  x1F   = (float*) alloc((size_t)B * H * 4);
    __bf16* x1B   = (__bf16*)alloc((size_t)B * H * 2);
    float*  t1F   = (float*) alloc((size_t)B * H * 4);
    __bf16* x2B   = (__bf16*)alloc((size_t)B * H * 2);
    float*  t2F   = (float*) alloc((size_t)B * H * 4);
    __bf16* x1bB  = (__bf16*)alloc((size_t)B * H * 2);
    float*  e1F   = (float*) alloc((size_t)B * H * 4);
    float*  e2F   = (float*) alloc((size_t)B * H * 4);
    int*    topId = (int*)   alloc((size_t)B * 50 * 4);

    auto blocks = [](size_t n) { return dim3((unsigned)((n + 255) / 256)); };
    auto gemm = [&](const __bf16* X, const __bf16* W, const float* bias, float* Y,
                    int M, int N, int K) {
        dim3 g((unsigned)((N + 63) / 64), (unsigned)(M / 128));
        gemm_bf16_kernel<<<g, dim3(256), 0, stream>>>(X, W, bias, Y, M, N, K);
    };

    // 1) weight conversion (bf16) and transposes to (N,K) layout
    cvt_bf16_kernel<<<blocks((size_t)G3 * EC), 256, 0, stream>>>(Wih, WihB, (size_t)G3 * EC);
    cvt_bf16_kernel<<<blocks((size_t)G3 * EC), 256, 0, stream>>>(Whh, WhhB, (size_t)G3 * EC);
    transpose_cvt_kernel<<<blocks((size_t)CIN * H), 256, 0, stream>>>(Wc, WcT, CIN, H);
    transpose_cvt_kernel<<<blocks((size_t)H * H), 256, 0, stream>>>(W1, W1T, H, H);
    transpose_cvt_kernel<<<blocks((size_t)H * H), 256, 0, stream>>>(W2, W2T, H, H);
    transpose_cvt_kernel<<<blocks((size_t)H * H), 256, 0, stream>>>(Wd1, Wd1T, H, H);
    transpose_cvt_kernel<<<blocks((size_t)H * H), 256, 0, stream>>>(Wd2, Wd2T, H, H);
    transpose_cvt_kernel<<<blocks((size_t)H * T), 256, 0, stream>>>(Wp, WpT, H, T);

    // 2) gather city embeddings; zero h0 (hF f32 zeros, hB bf16 zeros via f32-zero bits)
    gather_xc_kernel<<<blocks((size_t)5 * B * EC), 256, 0, stream>>>(x, emb_city, xcB);
    fill_f32_kernel<<<blocks((size_t)B * EC), 256, 0, stream>>>(hF, 0.f, (size_t)B * EC);
    fill_f32_kernel<<<blocks((size_t)B * EC / 2), 256, 0, stream>>>((float*)hB, 0.f, (size_t)B * EC / 2);

    // 3) GRU: 5 steps
    for (int t = 0; t < 5; ++t) {
        gemm(xcB + (size_t)t * B * EC, WihB, bih, gi, B, G3, EC);
        gemm(hB, WhhB, bhh, gh, B, G3, EC);
        gru_gate_kernel<<<blocks((size_t)B * EC), 256, 0, stream>>>(gi, gh, hF, hB);
    }

    // 4) feature concat (other embeddings + h), then x1 = feat @ Wc + bc
    feat_kernel<<<blocks((size_t)B * CIN), 256, 0, stream>>>(x, e0, e1, e2, e3, e4, e5, hF, featB);
    gemm(featB, WcT, bc, x1F, B, H, CIN);
    cvt_bf16_kernel<<<blocks((size_t)B * H), 256, 0, stream>>>(x1F, x1B, (size_t)B * H);

    // 5) x2 = relu(bn(x1 @ W1 + b1)); prob = x2 @ Wp + bp   (written into d_out[0:S])
    gemm(x1B, W1T, b1, t1F, B, H, H);
    bn_relu_kernel<<<dim3((H + 3) / 4), dim3(128), 0, stream>>>(t1F, g1, be1, x2B, H);
    gemm(x2B, WpT, bp, out, B, T, H);

    // 6) top-50 of prob
    topk_kernel<<<dim3(B), dim3(256), 0, stream>>>(out, topId, T);

    // 7) x1b = relu(bn(x1 @ W2 + b2)); e1 = x1b @ Wd1 + bd1; e2 = x2 @ Wd2 + bd2
    gemm(x1B, W2T, b2, t2F, B, H, H);
    bn_relu_kernel<<<dim3((H + 3) / 4), dim3(128), 0, stream>>>(t2F, g2, be2, x1bB, H);
    gemm(x1bB, Wd1T, bd1, e1F, B, H, H);
    gemm(x2B, Wd2T, bd2, e2F, B, H, H);

    // 8) prob_1 / prob_2: fill 1e-6, softmax-scatter
    fill_f32_kernel<<<blocks(2 * S), 256, 0, stream>>>(out + S, 1e-6f, 2 * S);
    emb_dot_kernel<<<dim3(B), dim3(64), 0, stream>>>(e1F, emb_city, topId, out + S, T);
    emb_dot_kernel<<<dim3(B), dim3(64), 0, stream>>>(e2F, emb_city, topId, out + 2 * S, T);

    // 9) weighted combine -> prob_ws
    combine_kernel<<<blocks(S), 256, 0, stream>>>(out, out + S, out + 2 * S, w1p, w2p,
                                                  out + 3 * S, S);
}
